// CuttleNet_49323404427548
// MI455X (gfx1250) — matrix operand order repaired and checked by reading, not compiled, not run on hardware
//
#include <hip/hip_runtime.h>

// ---------------------------------------------------------------------------
// CuttleNet forward for MI455X (gfx1250): batch-tiled WMMA f16 GEMM pipeline.
// v2: x tile staged coalesced into LDS (f16) once per wave; all gathers from
//     LDS. 2 waves / 64 threads per block to fit 64KB static LDS.
// ---------------------------------------------------------------------------

typedef __attribute__((ext_vector_type(16))) _Float16 v16h;
typedef __attribute__((ext_vector_type(8)))  _Float16 v8h;
typedef __attribute__((ext_vector_type(8)))  float    v8f;
typedef __attribute__((ext_vector_type(4)))  float    f32x4;

#define NLAYERS 34   // class_fc1, class_fc2, then 8 subnets x (fc1, h0, h1, fc2)
#define XSTRIDE 304  // padded LDS stride (halves) for the 16x300 x tile

struct Layer { int K, N, Kp, Np, KT, NT, wOff, bOff; };
struct Tab   { Layer L[NLAYERS]; int wTotal; int bTotal; };

static constexpr int ceil16(int x) { return (x + 15) & ~15; }
static constexpr int ceil32(int x) { return (x + 31) & ~31; }

static constexpr void add_layer(Tab& t, int& li, int& w, int& b, int K, int N) {
    Layer& L = t.L[li];
    L.K = K; L.N = N;
    L.Kp = ceil32(K); L.Np = ceil16(N);
    L.KT = L.Kp / 32; L.NT = L.Np / 16;
    L.wOff = w; L.bOff = b;
    w += L.KT * L.NT * 512;   // halves per layer (frag = 32 lanes * 16 halves)
    b += L.Np;                // padded bias floats
    ++li;
}

static constexpr Tab build_tab() {
    Tab t{};
    int S[8] = {40, 30, 15, 10, 5, 10, 19, 1};
    int w = 0, b = 0, li = 0;
    add_layer(t, li, w, b, 60, 16);   // class fc1
    add_layer(t, li, w, b, 16, 8);    // class fc2
    for (int c = 0; c < 8; ++c) {
        int twoS = 2 * S[c];
        add_layer(t, li, w, b, 108,  twoS);   // fc1
        add_layer(t, li, w, b, twoS, twoS);   // hidden0
        add_layer(t, li, w, b, twoS, twoS);   // hidden1
        add_layer(t, li, w, b, twoS, S[c]);   // fc2
    }
    t.wTotal = w; t.bTotal = b;
    return t;
}

static constexpr Tab h_tab = build_tab();
__constant__ Tab d_tab = build_tab();
__constant__ int d_colbase[8] = {0, 40, 70, 85, 95, 100, 110, 129};
__constant__ int d_S[8]       = {40, 30, 15, 10, 5, 10, 19, 1};

struct SrcPtrs { const float* p[2 * NLAYERS]; };   // [2*li]=W, [2*li+1]=b

// ---------------------------------------------------------------------------
// Prep kernel: pack f32 weights -> f16 fragment-major, pad biases.
// Fragment layout (matches load_b below): frag(nt,kt) is 512 contiguous halves
//   lane l holds 16 contiguous halves: n = nt*16 + (l&15),
//   k = kt*32 + ((l>>4)<<3) + (h&7) + ((h>>3)<<4)   for h = 0..15.
// ---------------------------------------------------------------------------
__global__ void prep_kernel(SrcPtrs sp, _Float16* __restrict__ wdst,
                            float* __restrict__ bdst) {
    int id = blockIdx.x * blockDim.x + threadIdx.x;
    if (id < d_tab.wTotal) {
        int li = 0;
        while (li + 1 < NLAYERS && id >= d_tab.L[li + 1].wOff) ++li;
        const Layer& L = d_tab.L[li];
        int e      = id - L.wOff;
        int fi     = e >> 9;
        int within = e & 511;
        int lane   = within >> 4;
        int h      = within & 15;
        int nt = fi / L.KT, kt = fi - nt * L.KT;
        int n = nt * 16 + (lane & 15);
        int k = kt * 32 + ((lane >> 4) << 3) + (h & 7) + ((h >> 3) << 4);
        float v = 0.0f;
        if (k < L.K && n < L.N) v = sp.p[2 * li][(size_t)k * L.N + n];
        wdst[id] = (_Float16)v;
    } else {
        int id2 = id - d_tab.wTotal;
        if (id2 < d_tab.bTotal) {
            int li = 0;
            while (li + 1 < NLAYERS && id2 >= d_tab.L[li + 1].bOff) ++li;
            const Layer& L = d_tab.L[li];
            int n = id2 - L.bOff;
            bdst[id2] = (n < L.N) ? sp.p[2 * li + 1][n] : 0.0f;
        }
    }
}

// ---------------------------------------------------------------------------
// WMMA helpers
// ---------------------------------------------------------------------------
__device__ __forceinline__ v8f wmma_f16(v16h a, v16h b, v8f c) {
    return __builtin_amdgcn_wmma_f32_16x16x32_f16(
        /*neg_a=*/false, a, /*neg_b=*/false, b,
        /*c_mod=*/(short)0, c, /*reuse_a=*/false, /*reuse_b=*/false);
}

// A: row-major 16 x stride f16 in LDS. ISA 16-bit A 16x32 layout:
//   lane l: M = l&15 ; halves 0..7 <- K = kt*32 + ((l&16)?8:0) + 0..7
//                       halves 8..15 <- same + 16
__device__ __forceinline__ v16h load_a(const _Float16* A, int stride, int kt,
                                       int lane) {
    const int m  = lane & 15;
    const int kb = (kt << 5) + ((lane & 16) >> 1);
    const _Float16* p = A + m * stride + kb;
    v8h lo = *(const v8h*)(p);
    v8h hi = *(const v8h*)(p + 16);
    return __builtin_shufflevector(lo, hi, 0, 1, 2, 3, 4, 5, 6, 7,
                                   8, 9, 10, 11, 12, 13, 14, 15);
}

// B fragment: 512 contiguous halves, lane gets its 16 contiguous halves.
__device__ __forceinline__ v16h load_b(const _Float16* Wp, int lane) {
    return *(const v16h*)(Wp + (size_t)lane * 16);
}

// D = relu(A x W + b) -> f16 LDS (row-major, stride Os)
__device__ __forceinline__ void gemm_relu(const _Float16* A, int As,
                                          const _Float16* Wp, const float* bias,
                                          int KT, int NT, _Float16* Out, int Os,
                                          int lane) {
    const int nl = lane & 15;
    const int mb = (lane >> 4) << 3;
    for (int nt = 0; nt < NT; ++nt) {
        v8f acc = {};
        const _Float16* Wn = Wp + (size_t)nt * KT * 512;
        for (int kt = 0; kt < KT; ++kt) {
            v16h a = load_a(A, As, kt, lane);
            v16h b = load_b(Wn + (size_t)kt * 512, lane);
            acc = wmma_f16(a, b, acc);
        }
        float bv = bias[nt * 16 + nl];
#pragma unroll
        for (int v = 0; v < 8; ++v) {
            float r = fmaxf(acc[v] + bv, 0.0f);
            Out[(mb + v) * Os + nt * 16 + nl] = (_Float16)r;
        }
    }
}

// D = A x W + b -> f32 LDS logits (row-major, stride Os)
__device__ __forceinline__ void gemm_f32(const _Float16* A, int As,
                                         const _Float16* Wp, const float* bias,
                                         int KT, int NT, float* Out, int Os,
                                         int lane) {
    const int nl = lane & 15;
    const int mb = (lane >> 4) << 3;
    for (int nt = 0; nt < NT; ++nt) {
        v8f acc = {};
        const _Float16* Wn = Wp + (size_t)nt * KT * 512;
        for (int kt = 0; kt < KT; ++kt) {
            v16h a = load_a(A, As, kt, lane);
            v16h b = load_b(Wn + (size_t)kt * 512, lane);
            acc = wmma_f16(a, b, acc);
        }
        float bv = bias[nt * 16 + nl];
#pragma unroll
        for (int v = 0; v < 8; ++v)
            Out[(mb + v) * Os + nt * 16 + nl] = acc[v] + bv;
    }
}

__device__ __forceinline__ void zero_lds(_Float16* p, int nhalves, int lane) {
    unsigned int* q = (unsigned int*)p;
    for (int j = lane; j < (nhalves >> 1); j += 32) q[j] = 0u;
}

// ---------------------------------------------------------------------------
// Forward kernel: 2 waves/block, each wave owns a 16-row tile end-to-end.
// x tile staged coalesced into LDS (f16) once; all gathers hit LDS.
// ---------------------------------------------------------------------------
__global__ __launch_bounds__(64) void fwd_kernel(
    const float* __restrict__ x, const int* __restrict__ cgi,
    const int* __restrict__ sgi, const _Float16* __restrict__ wts,
    const float* __restrict__ bias, float* __restrict__ out, int rows) {
    __shared__ __align__(16) _Float16 s_xt[2][16 * XSTRIDE]; // staged x tile
    __shared__ __align__(16) _Float16 s_ab[2][16 * 128];     // A buf (+f32 alias)
    __shared__ __align__(16) _Float16 s_a0[2][16 * 96];      // fc1 out (= skip)
    __shared__ __align__(16) _Float16 s_a1[2][16 * 96];
    __shared__ __align__(16) _Float16 s_a2[2][16 * 96];
    __shared__ __align__(16) _Float16 s_ch[2][16 * 32];      // class hidden
    __shared__ float s_co[2][16 * 8];                        // class log-probs
    __shared__ float s_lse[2][16];

    const int lane = threadIdx.x & 31;
    const int w    = threadIdx.x >> 5;
    const size_t m0 = ((size_t)blockIdx.x * 2 + w) * 16;
    _Float16* ab = s_ab[w];
    _Float16* xt = s_xt[w];
    float* sbuf  = (float*)ab;   // f32 logits alias (A not needed then)

    // ---------------- stage x tile: coalesced global -> LDS f16 ------------
    if (m0 + 16 <= (size_t)rows) {
        // fast path: flat float4 loads (x + m0*300 is 16B aligned: 300*4*m0)
        const f32x4* xb = (const f32x4*)(x + m0 * 300);
        for (int j = lane; j < (16 * 300) / 4; j += 32) {
            f32x4 v = xb[j];
            int f = j * 4;
#pragma unroll
            for (int e = 0; e < 4; ++e) {
                int fe = f + e;
                int r = fe / 300, col = fe - r * 300;
                xt[r * XSTRIDE + col] = (_Float16)v[e];
            }
        }
    } else {
        for (int j = lane; j < 16 * 300; j += 32) {
            int r = j / 300, col = j - r * 300;
            size_t rr = m0 + r; if (rr >= (size_t)rows) rr = rows - 1;
            xt[r * XSTRIDE + col] = (_Float16)x[rr * 300 + col];
        }
    }
    zero_lds(s_ch[w], 16 * 32, lane);
    __syncthreads();

    // ---------------- class head ----------------
    for (int j = lane; j < 16 * 64; j += 32) {
        int r = j >> 6, k = j & 63;
        ab[j] = (k < 60) ? xt[r * XSTRIDE + cgi[k]] : (_Float16)0.0f;
    }
    __syncthreads();

    {   // class fc1: 16x64 * 64x16, relu -> s_ch (stride 32, upper cols zero)
        const Layer& L = d_tab.L[0];
        gemm_relu(ab, 64, wts + L.wOff, bias + L.bOff, L.KT, L.NT,
                  s_ch[w], 32, lane);
    }
    __syncthreads();
    {   // class fc2 -> f32 logits in sbuf (stride 16)
        const Layer& L = d_tab.L[1];
        gemm_f32(s_ch[w], 32, wts + L.wOff, bias + L.bOff, L.KT, L.NT,
                 sbuf, 16, lane);
    }
    __syncthreads();
    if (lane < 16) {   // log-sum-exp over 8 classes per row
        const float* row = sbuf + lane * 16;
        float mx = row[0];
#pragma unroll
        for (int s = 1; s < 8; ++s) mx = fmaxf(mx, row[s]);
        float sum = 0.0f;
#pragma unroll
        for (int s = 0; s < 8; ++s) sum += __expf(row[s] - mx);
        s_lse[w][lane] = mx + __logf(sum);
    }
    __syncthreads();
    for (int j = lane; j < 16 * 8; j += 32) {
        int r = j >> 3, n = j & 7;
        s_co[w][j] = sbuf[r * 16 + n] - s_lse[w][r];
    }
    __syncthreads();

    // ---------------- subnets ----------------
    for (int c = 0; c < 8; ++c) {
        const Layer& L1  = d_tab.L[2 + c * 4 + 0];
        const Layer& Lh0 = d_tab.L[2 + c * 4 + 1];
        const Layer& Lh1 = d_tab.L[2 + c * 4 + 2];
        const Layer& L2  = d_tab.L[2 + c * 4 + 3];
        const int KH = Lh0.Kp;   // activation stride (padded 2S)
        const int S  = d_S[c];

        // build A = [gathered 100 genes | class logprobs 8 | zeros] (16 x 128)
        for (int j = lane; j < 16 * 128; j += 32) {
            int r = j >> 7, k = j & 127;
            _Float16 v;
            if (k < 100)      v = xt[r * XSTRIDE + sgi[c * 100 + k]];
            else if (k < 108) v = (_Float16)s_co[w][r * 8 + (k - 100)];
            else              v = (_Float16)0.0f;
            ab[j] = v;
        }
        zero_lds(s_a0[w], 16 * 96, lane);
        zero_lds(s_a1[w], 16 * 96, lane);
        zero_lds(s_a2[w], 16 * 96, lane);
        __syncthreads();

        gemm_relu(ab, 128, wts + L1.wOff, bias + L1.bOff, L1.KT, L1.NT,
                  s_a0[w], KH, lane);
        __syncthreads();
        gemm_relu(s_a0[w], KH, wts + Lh0.wOff, bias + Lh0.bOff, Lh0.KT, Lh0.NT,
                  s_a1[w], KH, lane);
        __syncthreads();
        gemm_relu(s_a1[w], KH, wts + Lh1.wOff, bias + Lh1.bOff, Lh1.KT, Lh1.NT,
                  s_a2[w], KH, lane);
        __syncthreads();
        // skip connection: h = relu(h2) + fc1_out
        for (int j = lane; j < 16 * KH; j += 32)
            s_a2[w][j] = (_Float16)((float)s_a2[w][j] + (float)s_a0[w][j]);
        __syncthreads();
        // fc2 -> f32 logits (stride L2.Np), padded cols are exact zeros
        gemm_f32(s_a2[w], KH, wts + L2.wOff, bias + L2.bOff, L2.KT, L2.NT,
                 sbuf, L2.Np, lane);
        __syncthreads();
        if (lane < 16) {
            const float* row = sbuf + lane * L2.Np;
            float mx = -3.4e38f;
            for (int s = 0; s < S; ++s) mx = fmaxf(mx, row[s]);
            float sum = 0.0f;
            for (int s = 0; s < S; ++s) sum += __expf(row[s] - mx);
            s_lse[w][lane] = mx + __logf(sum);
        }
        __syncthreads();
        const int cb = d_colbase[c];
        for (int j = lane; j < 16 * S; j += 32) {
            int r = j / S, s = j - r * S;
            size_t rr = m0 + r;
            if (rr < (size_t)rows) {
                float lp = sbuf[r * L2.Np + s] - s_lse[w][r];
                out[rr * 130 + cb + s] = lp * s_co[w][r * 8 + c];
            }
        }
        __syncthreads();
    }
}

// ---------------------------------------------------------------------------
// Launch
// ---------------------------------------------------------------------------
extern "C" void kernel_launch(void* const* d_in, const int* in_sizes, int n_in,
                              void* d_out, int out_size, void* d_ws,
                              size_t ws_size, hipStream_t stream) {
    const float* x  = (const float*)d_in[0];
    const int* cgi  = (const int*)d_in[1];
    const int* sgi  = (const int*)d_in[2];

    // params in setup_inputs() insertion order:
    // class_fc1 (W,b), class_fc2 (W,b), then per subnet: fc1(W,b),
    // hidden0(W,b), hidden1(W,b), fc2(W,b)
    SrcPtrs sp{};
    sp.p[0] = (const float*)d_in[3];  sp.p[1] = (const float*)d_in[4];
    sp.p[2] = (const float*)d_in[5];  sp.p[3] = (const float*)d_in[6];
    for (int c = 0; c < 8; ++c) {
        int base = 7 + c * 8;
        int li   = 2 + c * 4;
        for (int k = 0; k < 8; ++k)
            sp.p[2 * li + k] = (const float*)d_in[base + k];
    }

    _Float16* wts = (_Float16*)d_ws;
    float* bb = (float*)((char*)d_ws + (size_t)h_tab.wTotal * sizeof(_Float16));

    int total = h_tab.wTotal + h_tab.bTotal;
    prep_kernel<<<(total + 255) / 256, 256, 0, stream>>>(sp, wts, bb);

    size_t rows = (size_t)in_sizes[0] / 300;
    int blocks  = (int)((rows + 31) / 32);
    fwd_kernel<<<blocks, 64, 0, stream>>>(x, cgi, sgi, wts, bb, (float*)d_out,
                                          (int)rows);
}
